// IterativeRefinementBlock_76579266888372
// MI455X (gfx1250) — matrix operand order repaired and checked by reading, not compiled, not run on hardware
//
#include <hip/hip_runtime.h>
#include <hip/hip_bf16.h>
#include <math.h>

// ---------------------------------------------------------------------------
// Problem constants (reference: B=4, L=2048, D=256, P=32, NIT=3)
// ---------------------------------------------------------------------------
#define BB   4
#define LL   2048
#define DD   256
#define PP   32
#define NIT  3
#define TT   (BB * LL)          // 8192 tokens
#define PI_F 3.14159265358979323846f

typedef __attribute__((ext_vector_type(16))) __bf16 bf16x16;
typedef __attribute__((ext_vector_type(8)))  __bf16 bf16x8;
typedef __attribute__((ext_vector_type(2)))  __bf16 bf16x2;
typedef __attribute__((ext_vector_type(8)))  float  f32x8;

// ---------------------------------------------------------------------------
// WMMA helpers. CDNA5 v_wmma_f32_16x16x32_bf16, wave32.
// A-frag (16x32, MxK): lane = 16h + m ; elem j -> k = 16h + j   (contiguous)
// B-frag (32x16, KxN): lane = 16h + n ; elem j -> k = 16h + j   (contiguous)
// C/D    (16x16)     : lane = 16h + n ; vgpr r -> m = r + 8h
// ---------------------------------------------------------------------------
__device__ __forceinline__ f32x8 wmma_bf16(bf16x16 a, bf16x16 b, f32x8 c) {
    return __builtin_amdgcn_wmma_f32_16x16x32_bf16(
        /*neg_a=*/false, a, /*neg_b=*/false, b,
        /*c_mod=*/(short)0, c, /*reuse_a=*/false, /*reuse_b=*/false);
}

// row points at element k=0 of this lane's M-row of a 32-wide k chunk.
__device__ __forceinline__ bf16x16 load_a_frag(const __bf16* row, int h) {
    bf16x8 lo = *(const bf16x8*)(row + 16 * h);
    bf16x8 hi = *(const bf16x8*)(row + 16 * h + 8);
    bf16x16 r;
#pragma unroll
    for (int j = 0; j < 8; ++j) { r[j] = lo[j]; r[j + 8] = hi[j]; }
    return r;
}

// ---------------------------------------------------------------------------
// Elementwise kernels
// ---------------------------------------------------------------------------
__global__ void cvt_f32_bf16_kernel(const float* __restrict__ src,
                                    __bf16* __restrict__ dst, int n) {
    int i = blockIdx.x * blockDim.x + threadIdx.x;
    if (i < n) dst[i] = (__bf16)src[i];
}

__global__ void init_state_kernel(const float* __restrict__ x,
                                  __bf16* __restrict__ xb,
                                  float* __restrict__ qf,
                                  __bf16* __restrict__ qb,
                                  float* __restrict__ accum, int n) {
    int i = blockIdx.x * blockDim.x + threadIdx.x;
    if (i < n) {
        float v = x[i];
        xb[i] = (__bf16)v;
        qf[i] = v;
        qb[i] = (__bf16)v;
        accum[i] = 0.0f;
    }
}

// ---------------------------------------------------------------------------
// LayerNorm: one wave per token (D=256 -> 8 f32 per lane), bf16 output.
// ---------------------------------------------------------------------------
__global__ __launch_bounds__(256)
void ln_kernel(const float* __restrict__ in, const float* __restrict__ g,
               const float* __restrict__ b, __bf16* __restrict__ out) {
    int wid = threadIdx.x >> 5;
    int lane = threadIdx.x & 31;
    int tok = blockIdx.x * 8 + wid;
    const float* row = in + (size_t)tok * DD;
    float v[8];
    float s = 0.f, ss = 0.f;
#pragma unroll
    for (int j = 0; j < 8; ++j) {
        v[j] = row[lane * 8 + j];
        s += v[j];
        ss += v[j] * v[j];
    }
#pragma unroll
    for (int o = 16; o >= 1; o >>= 1) {
        s  += __shfl_xor(s,  o, 32);
        ss += __shfl_xor(ss, o, 32);
    }
    float mean = s * (1.0f / DD);
    float var  = ss * (1.0f / DD) - mean * mean;
    float inv  = rsqrtf(var + 1e-5f);
    __bf16* orow = out + (size_t)tok * DD;
#pragma unroll
    for (int j = 0; j < 8; ++j) {
        int c = lane * 8 + j;
        orow[c] = (__bf16)((v[j] - mean) * inv * g[c] + b[c]);
    }
}

// ---------------------------------------------------------------------------
// Generic token-parallel WMMA GEMM with fused epilogues.
// Block: 128 threads = 4 waves. Tile: 64 tokens x NT cols. K in chunks of 32.
// Register-pipelined staging: global loads for tile k+1 are issued before
// the WMMA block for tile k, and only stored to LDS at the next iteration.
// ---------------------------------------------------------------------------
enum { EP_BIAS_BF16 = 0, EP_PHASOR = 1, EP_GELU = 2,
       EP_REFINE = 3, EP_GATE = 4, EP_FINAL = 5 };

template <int NT, int EP>
__global__ __launch_bounds__(128)
void gemm_ep_kernel(const __bf16* __restrict__ A0, int K0,
                    const __bf16* __restrict__ A1, int K1,
                    const __bf16* __restrict__ W, int ldw,
                    const float* __restrict__ bias,
                    float* __restrict__ of32,
                    __bf16* __restrict__ obf16,
                    const float* __restrict__ aux,
                    int ldo) {
    __shared__ __bf16 sA[64 * 32];
    __shared__ __bf16 sW[NT * 32];   // transposed: [n][k]

    const int tid  = threadIdx.x;
    const int lane = tid & 31;
    const int w    = tid >> 5;
    const int h    = lane >> 4;
    const int nl   = lane & 15;
    const int row0 = blockIdx.x * 64;
    const int n0   = blockIdx.y * NT;
    const int K    = K0 + K1;

    // staging indices (fixed per thread)
    const int ar = tid >> 1;
    const int ac = (tid & 1) * 16;
    const int wkr = (tid / (NT / 8)) * 2;   // valid when tid < 2*NT
    const int wnb = (tid % (NT / 8)) * 8;

    bf16x8 ra0, ra1, rw0, rw1;
    auto load_tile = [&](int kk) {
        int kg = kk + ac;
        const __bf16* src = (kg < K0)
            ? (A0 + (size_t)(row0 + ar) * K0 + kg)
            : (A1 + (size_t)(row0 + ar) * K1 + (kg - K0));
        ra0 = *(const bf16x8*)src;
        ra1 = *(const bf16x8*)(src + 8);
        if (tid < 2 * NT) {
            rw0 = *(const bf16x8*)&W[(size_t)(kk + wkr) * ldw + n0 + wnb];
            rw1 = *(const bf16x8*)&W[(size_t)(kk + wkr + 1) * ldw + n0 + wnb];
        }
    };
    auto store_tile = [&]() {
        *(bf16x8*)&sA[ar * 32 + ac]     = ra0;
        *(bf16x8*)&sA[ar * 32 + ac + 8] = ra1;
        if (tid < 2 * NT) {
#pragma unroll
            for (int j = 0; j < 8; ++j) {
                bf16x2 p; p[0] = rw0[j]; p[1] = rw1[j];
                *(bf16x2*)&sW[(wnb + j) * 32 + wkr] = p;
            }
        }
    };

    f32x8 acc[NT / 16];
#pragma unroll
    for (int nn = 0; nn < NT / 16; ++nn)
#pragma unroll
        for (int j = 0; j < 8; ++j) acc[nn][j] = 0.0f;

    load_tile(0);
    for (int kk = 0; kk < K; kk += 32) {
        store_tile();
        __syncthreads();
        if (kk + 32 < K) load_tile(kk + 32);   // overlaps with WMMAs below

        bf16x16 a = load_a_frag(&sA[(w * 16 + nl) * 32], h);
#pragma unroll
        for (int nn = 0; nn < NT / 16; ++nn) {
            bf16x16 bfr = *(const bf16x16*)&sW[(nn * 16 + nl) * 32 + 16 * h];
            acc[nn] = wmma_bf16(a, bfr, acc[nn]);
        }
        __syncthreads();
    }

    // epilogue (lane nl = col within 16-chunk; vgpr r -> token row)
#pragma unroll
    for (int nn = 0; nn < NT / 16; ++nn) {
#pragma unroll
        for (int r = 0; r < 8; ++r) {
            int m   = r + 8 * h;
            int tok = row0 + w * 16 + m;
            int col = n0 + nn * 16 + nl;
            float s = acc[nn][r];
            if (EP == EP_BIAS_BF16) {
                obf16[(size_t)tok * ldo + col] = (__bf16)(s + bias[col]);
            } else if (EP == EP_PHASOR) {
                float ph = tanhf(s + bias[col]) * PI_F;
                float sn, cs;
                sincosf(ph, &sn, &cs);
                obf16[(size_t)tok * 64 + col]      = (__bf16)cs;
                obf16[(size_t)tok * 64 + 32 + col] = (__bf16)sn;
            } else if (EP == EP_GELU) {
                float v = s + bias[col];
                float gl = 0.5f * v * (1.0f + erff(v * 0.70710678118f));
                obf16[(size_t)tok * ldo + col] = (__bf16)gl;
            } else if (EP == EP_REFINE) {
                float v = s + bias[col];
                of32[(size_t)tok * ldo + col] += v;          // accumulated +=
                obf16[(size_t)tok * ldo + col] = (__bf16)v;  // refined
            } else if (EP == EP_GATE) {
                float gt = tanhf(s + bias[col]);
                float nq = aux[(size_t)tok * ldo + col] + gt;
                of32[(size_t)tok * ldo + col]  = nq;
                obf16[(size_t)tok * ldo + col] = (__bf16)nq;
            } else { // EP_FINAL
                of32[(size_t)tok * ldo + col] =
                    aux[(size_t)tok * ldo + col] + s + bias[col];
            }
        }
    }
}

// ---------------------------------------------------------------------------
// Causal phase-correlation attention (no softmax):
//   out[l,d] = ( sum_{t<=l} (q_l . k_t) * v[t,d] ) / sqrt((l+1)*P)
//
// Transposed formulation: S^T = K Q^T (D rows = t -> packed b32 LDS stores
// into sS[m][t] which is exactly the B-frag layout for the next product),
// then O^T = V^T S^T (D rows = d -> per-lane 8 contiguous floats of one
// token row -> b128 global stores, one rsqrt per lane).
// Register-pipelined K/V staging: loads for key-tile kt+1 issue before the
// WMMA block for tile kt.
// Block: 128 threads = 4 waves; 64 query rows; 32-key tiles.
// ---------------------------------------------------------------------------
__global__ __launch_bounds__(128)
void attn_kernel(const __bf16* __restrict__ qph,
                 const __bf16* __restrict__ kph,
                 const __bf16* __restrict__ vbf,
                 float* __restrict__ out) {
    __shared__ __bf16 sQ[64 * 64];     // [m][c]
    __shared__ __bf16 sK[32 * 64];     // [t][c]
    __shared__ __bf16 sVt[256 * 32];   // transposed: [d][t]
    __shared__ __bf16 sS[4][16 * 32];  // per-wave scores [m][t]

    const int tid  = threadIdx.x;
    const int lane = tid & 31;
    const int w    = tid >> 5;
    const int h    = lane >> 4;
    const int nl   = lane & 15;
    const int b    = blockIdx.y;
    const int m0   = blockIdx.x * 64;
    const size_t base = (size_t)b * LL;

    // staged K/V tile in registers (one set; regs die at the LDS store)
    bf16x8 rk[2], rv[8];
    auto load_kv = [&](int kt) {
#pragma unroll
        for (int i = 0; i < 2; ++i) {
            int e = tid + i * 128;
            int r = e >> 3, c = (e & 7) * 8;
            rk[i] = *(const bf16x8*)&kph[(base + kt * 32 + r) * 64 + c];
        }
#pragma unroll
        for (int i = 0; i < 4; ++i) {
            int e = tid + i * 128;
            int t = (e >> 5) * 2, d0 = (e & 31) * 8;
            rv[2 * i]     = *(const bf16x8*)&vbf[(base + kt * 32 + t) * 256 + d0];
            rv[2 * i + 1] = *(const bf16x8*)&vbf[(base + kt * 32 + t + 1) * 256 + d0];
        }
    };
    auto store_kv = [&]() {
#pragma unroll
        for (int i = 0; i < 2; ++i) {
            int e = tid + i * 128;
            int r = e >> 3, c = (e & 7) * 8;
            *(bf16x8*)&sK[r * 64 + c] = rk[i];
        }
#pragma unroll
        for (int i = 0; i < 4; ++i) {
            int e = tid + i * 128;
            int t = (e >> 5) * 2, d0 = (e & 31) * 8;
#pragma unroll
            for (int j = 0; j < 8; ++j) {
                bf16x2 p; p[0] = rv[2 * i][j]; p[1] = rv[2 * i + 1][j];
                *(bf16x2*)&sVt[(d0 + j) * 32 + t] = p;
            }
        }
    };

    load_kv(0);   // in flight while we stage Q

    // stage Q tile (64 x 64)
    for (int e = tid; e < 64 * 64 / 8; e += 128) {
        int r = e >> 3;
        int c = (e & 7) * 8;
        *(bf16x8*)&sQ[r * 64 + c] =
            *(const bf16x8*)&qph[(base + m0 + r) * 64 + c];
    }
    __syncthreads();

    // Q^T as B-frags: B[k=c][n=m], lane = 16h + m, elem j -> c = cc*32+16h+j
    bf16x16 qb[2];
#pragma unroll
    for (int cc = 0; cc < 2; ++cc)
        qb[cc] = *(const bf16x16*)&sQ[(w * 16 + nl) * 64 + cc * 32 + 16 * h];

    f32x8 acc[16];   // O^T: frag dd -> rows d = dd*16 + r + 8h, col m = nl
#pragma unroll
    for (int i = 0; i < 16; ++i)
#pragma unroll
        for (int j = 0; j < 8; ++j) acc[i][j] = 0.0f;

    const int mg  = m0 + w * 16 + nl;   // this lane's query index (fixed)
    const int nkt = (m0 >> 5) + 2;      // key tiles through the diagonal
    for (int kt = 0; kt < nkt; ++kt) {
        store_kv();
        __syncthreads();
        if (kt + 1 < nkt) load_kv(kt + 1);   // overlaps with WMMAs below

        // S^T tile: 32(t) x 16(m) per wave as two 16x16 frags
        const bool full = (kt * 32 + 31) <= (m0 + w * 16);  // wave-uniform
#pragma unroll
        for (int tt = 0; tt < 2; ++tt) {
            f32x8 s;
#pragma unroll
            for (int j = 0; j < 8; ++j) s[j] = 0.0f;
#pragma unroll
            for (int cc = 0; cc < 2; ++cc) {
                bf16x16 ka = load_a_frag(&sK[(tt * 16 + nl) * 64 + cc * 32], h);
                s = wmma_bf16(ka, qb[cc], s);
            }
            // causal mask (only diagonal tiles) + packed b32 store to [m][t]
#pragma unroll
            for (int r = 0; r < 8; r += 2) {
                float v0 = s[r], v1 = s[r + 1];
                if (!full) {
                    int tg = kt * 32 + tt * 16 + 8 * h + r;
                    v0 = (tg     <= mg) ? v0 : 0.0f;
                    v1 = (tg + 1 <= mg) ? v1 : 0.0f;
                }
                bf16x2 p; p[0] = (__bf16)v0; p[1] = (__bf16)v1;
                *(bf16x2*)&sS[w][nl * 32 + tt * 16 + 8 * h + r] = p;
            }
        }
        // O^T += V^T * S^T ; sS[m][t] is already the B-frag layout
        bf16x16 sb = *(const bf16x16*)&sS[w][nl * 32 + 16 * h];
#pragma unroll
        for (int dd = 0; dd < 16; ++dd) {
            bf16x16 va = load_a_frag(&sVt[(dd * 16 + nl) * 32], h);
            acc[dd] = wmma_bf16(va, sb, acc[dd]);
        }
        __syncthreads();
    }

    // store: lane owns one token row; 8 contiguous d per frag -> b128 stores
    const int l = m0 + w * 16 + nl;
    const float scale = rsqrtf((float)(l + 1) * (float)PP);
    float* orow = out + (base + l) * DD;
#pragma unroll
    for (int dd = 0; dd < 16; ++dd) {
        float4 o0, o1;
        o0.x = acc[dd][0] * scale; o0.y = acc[dd][1] * scale;
        o0.z = acc[dd][2] * scale; o0.w = acc[dd][3] * scale;
        o1.x = acc[dd][4] * scale; o1.y = acc[dd][5] * scale;
        o1.z = acc[dd][6] * scale; o1.w = acc[dd][7] * scale;
        *(float4*)&orow[dd * 16 + 8 * h]     = o0;
        *(float4*)&orow[dd * 16 + 8 * h + 4] = o1;
    }
}

// ---------------------------------------------------------------------------
// Host side
// ---------------------------------------------------------------------------
extern "C" void kernel_launch(void* const* d_in, const int* in_sizes, int n_in,
                              void* d_out, int out_size, void* d_ws,
                              size_t ws_size, hipStream_t stream) {
    (void)in_sizes; (void)n_in; (void)out_size; (void)ws_size;

    const float* x        = (const float*)d_in[0];
    const float* pe_w     = (const float*)d_in[1];
    const float* pe_b     = (const float*)d_in[2];
    const float* tv_w     = (const float*)d_in[3];
    const float* tv_b     = (const float*)d_in[4];
    const float* ref_ln_g = (const float*)d_in[5];
    const float* ref_ln_b = (const float*)d_in[6];
    const float* ref_w1   = (const float*)d_in[7];
    const float* ref_b1   = (const float*)d_in[8];
    const float* ref_w2   = (const float*)d_in[9];
    const float* ref_b2   = (const float*)d_in[10];
    const float* qg_w     = (const float*)d_in[11];
    const float* qg_b     = (const float*)d_in[12];
    const float* out_ln_g = (const float*)d_in[13];
    const float* out_ln_b = (const float*)d_in[14];
    const float* out_w    = (const float*)d_in[15];
    const float* out_b    = (const float*)d_in[16];
    float* out            = (float*)d_out;

    // workspace carve-up (256B aligned)
    char* base = (char*)d_ws;
    size_t off = 0;
    auto alloc = [&](size_t bytes) -> char* {
        char* p = base + off;
        off = (off + bytes + 255) & ~(size_t)255;
        return p;
    };
    __bf16* wpe  = (__bf16*)alloc((size_t)DD * PP * 2);
    __bf16* wtv  = (__bf16*)alloc((size_t)DD * DD * 2);
    __bf16* w1b  = (__bf16*)alloc((size_t)NIT * DD * 2 * DD * 2);
    __bf16* w2b  = (__bf16*)alloc((size_t)NIT * 2 * DD * DD * 2);
    __bf16* wqg  = (__bf16*)alloc((size_t)NIT * 2 * DD * DD * 2);
    __bf16* wout = (__bf16*)alloc((size_t)DD * DD * 2);

    __bf16* xb      = (__bf16*)alloc((size_t)TT * DD * 2);
    __bf16* qbf0    = (__bf16*)alloc((size_t)TT * DD * 2);
    __bf16* qbf1    = (__bf16*)alloc((size_t)TT * DD * 2);
    float*  qf0     = (float*)alloc((size_t)TT * DD * 4);
    float*  qf1     = (float*)alloc((size_t)TT * DD * 4);
    __bf16* vbf     = (__bf16*)alloc((size_t)TT * DD * 2);
    __bf16* kph     = (__bf16*)alloc((size_t)TT * 64 * 2);
    __bf16* qph     = (__bf16*)alloc((size_t)TT * 64 * 2);
    float*  retr    = (float*)alloc((size_t)TT * DD * 4);
    __bf16* hbf     = (__bf16*)alloc((size_t)TT * DD * 2);
    __bf16* h2bf    = (__bf16*)alloc((size_t)TT * 2 * DD * 2);
    __bf16* refb    = (__bf16*)alloc((size_t)TT * DD * 2);
    float*  accum   = (float*)alloc((size_t)TT * DD * 4);
    __bf16* alnb    = (__bf16*)alloc((size_t)TT * DD * 2);

    __bf16* qbf_buf[2] = { qbf0, qbf1 };
    float*  qf_buf[2]  = { qf0,  qf1  };

    auto cvt = [&](const float* s, __bf16* d, int n) {
        cvt_f32_bf16_kernel<<<(n + 255) / 256, 256, 0, stream>>>(s, d, n);
    };
    cvt(pe_w,   wpe,  DD * PP);
    cvt(tv_w,   wtv,  DD * DD);
    cvt(ref_w1, w1b,  NIT * DD * 2 * DD);
    cvt(ref_w2, w2b,  NIT * 2 * DD * DD);
    cvt(qg_w,   wqg,  NIT * 2 * DD * DD);
    cvt(out_w,  wout, DD * DD);

    init_state_kernel<<<(TT * DD + 255) / 256, 256, 0, stream>>>(
        x, xb, qf0, qbf0, accum, TT * DD);

    // values = x @ tv_w + tv_b  (bf16)
    gemm_ep_kernel<64, EP_BIAS_BF16><<<dim3(TT / 64, DD / 64), 128, 0, stream>>>(
        xb, DD, nullptr, 0, wtv, DD, tv_b, nullptr, vbf, nullptr, DD);

    // key phasors from x
    gemm_ep_kernel<32, EP_PHASOR><<<dim3(TT / 64, 1), 128, 0, stream>>>(
        xb, DD, nullptr, 0, wpe, PP, pe_b, nullptr, kph, nullptr, 64);

    for (int i = 0; i < NIT; ++i) {
        __bf16* qb_in  = qbf_buf[i & 1];
        float*  qf_in  = qf_buf[i & 1];
        __bf16* qb_out = qbf_buf[(i + 1) & 1];
        float*  qf_out = qf_buf[(i + 1) & 1];

        // query phasors
        gemm_ep_kernel<32, EP_PHASOR><<<dim3(TT / 64, 1), 128, 0, stream>>>(
            qb_in, DD, nullptr, 0, wpe, PP, pe_b, nullptr, qph, nullptr, 64);

        // causal phase-correlation retrieval
        attn_kernel<<<dim3(LL / 64, BB), 128, 0, stream>>>(qph, kph, vbf, retr);

        // LayerNorm -> h (bf16)
        ln_kernel<<<TT / 8, 256, 0, stream>>>(
            retr, ref_ln_g + i * DD, ref_ln_b + i * DD, hbf);

        // h @ w1 + b1 -> gelu -> h2 (bf16, 512 cols)
        gemm_ep_kernel<64, EP_GELU><<<dim3(TT / 64, 2 * DD / 64), 128, 0, stream>>>(
            hbf, DD, nullptr, 0, w1b + (size_t)i * DD * 2 * DD, 2 * DD,
            ref_b1 + i * 2 * DD, nullptr, h2bf, nullptr, 2 * DD);

        // h2 @ w2 + b2 -> refined; accum += refined
        gemm_ep_kernel<64, EP_REFINE><<<dim3(TT / 64, DD / 64), 128, 0, stream>>>(
            h2bf, 2 * DD, nullptr, 0, w2b + (size_t)i * 2 * DD * DD, DD,
            ref_b2 + i * DD, accum, refb, nullptr, DD);

        // gate: query_new = query + tanh([query, refined] @ qg_w + qg_b)
        gemm_ep_kernel<64, EP_GATE><<<dim3(TT / 64, DD / 64), 128, 0, stream>>>(
            qb_in, DD, refb, DD, wqg + (size_t)i * 2 * DD * DD, DD,
            qg_b + i * DD, qf_out, qb_out, qf_in, DD);
    }

    // out = x + LN(accum) @ out_w + out_b
    ln_kernel<<<TT / 8, 256, 0, stream>>>(accum, out_ln_g, out_ln_b, alnb);
    gemm_ep_kernel<64, EP_FINAL><<<dim3(TT / 64, DD / 64), 128, 0, stream>>>(
        alnb, DD, nullptr, 0, wout, DD, out_b, out, nullptr, x, DD);
}
